// EnhancedCNNCCBiMamba_78314433675556
// MI455X (gfx1250) — matrix operand order repaired and checked by reading, not compiled, not run on hardware
//
#include <hip/hip_runtime.h>
#include <hip/hip_bf16.h>
#include <math.h>

// ---------------------------------------------------------------------------
// MI455X (gfx1250) implementation notes:
//  * wave32 only; 256-thread blocks = 8 waves; block tile 128(M) x 64(N),
//    each wave owns a 32x32 sub-tile => 4x v_wmma_f32_16x16x32_bf16 per K-step
//  * LDS tiles double-buffered and filled by the Tensor Data Mover;
//    s_wait_tensorcnt(2) keeps the next tile's pair of TDM ops in flight
//    while the current tile feeds the WMMA pipe
//  * selective scan = register-resident recurrence, one lane per (batch, chan)
// ---------------------------------------------------------------------------

typedef __attribute__((ext_vector_type(16))) __bf16 v16bf;
typedef __attribute__((ext_vector_type(8)))  float  v8f;
typedef __attribute__((ext_vector_type(4)))  unsigned int v4u;
typedef __attribute__((ext_vector_type(8)))  int v8i;
typedef __attribute__((ext_vector_type(4)))  int v4i;

// -------------------------- device helpers ---------------------------------

template <int ACT>
__device__ __forceinline__ float act_apply(float v) {
  if (ACT == 1) return v / (1.f + __expf(-v));                       // SiLU
  if (ACT == 2) return 0.5f * v * (1.f + erff(v * 0.70710678118f));  // GELU (exact)
  if (ACT == 3) return (v > 20.f) ? v : log1pf(__expf(v));           // softplus
  return v;
}

// Build a TDM D# for a 2D [tile_rows x 32] bf16 tile and issue
// tensor_load_to_lds. LDS destination is padded: 64B of data then 16B pad per
// row (pad_interval code 3 = 16 DWORDs, pad_amount code 3 = 4 DWORDs) which
// matches a __shared__ __bf16 [rows][40] tile.
__device__ __forceinline__ void tdm_load_tile(unsigned lds_off, const void* gptr,
                                              int rows_avail, int cols_avail,
                                              long long row_stride_elems,
                                              unsigned tile_rows) {
  unsigned long long ga = (unsigned long long)(size_t)gptr;
  unsigned td0 = (unsigned)(cols_avail < 0 ? 0 : cols_avail);
  unsigned td1 = (unsigned)(rows_avail < 0 ? 0 : rows_avail);
  unsigned long long s0 = (unsigned long long)row_stride_elems;

  v4u g0;
  g0.x = 1u;                                               // count=1, user D#
  g0.y = lds_off;                                          // lds_addr
  g0.z = (unsigned)(ga & 0xFFFFFFFFu);                     // global_addr[31:0]
  g0.w = (unsigned)((ga >> 32) & 0x1FFFFFFu) | (2u << 30); // global_addr[56:32], type=2

  v8i g1;
  g1[0] = (int)((1u << 16) | (1u << 20) | (3u << 22) | (3u << 25)); // 2B data, pad en, 64B interval, 16B pad
  g1[1] = (int)((td0 & 0xFFFFu) << 16);                             // tensor_dim0[15:0]
  g1[2] = (int)(((td0 >> 16) & 0xFFFFu) | ((td1 & 0xFFFFu) << 16)); // dim0 hi | dim1 lo
  g1[3] = (int)(((td1 >> 16) & 0xFFFFu) | (32u << 16));             // dim1 hi | tile_dim0=32
  g1[4] = (int)tile_rows;                                           // tile_dim1, tile_dim2=0
  g1[5] = (int)(s0 & 0xFFFFFFFFull);                                // dim0_stride lo
  g1[6] = (int)((s0 >> 32) & 0xFFFFull);                            // dim0_stride hi
  g1[7] = 0;

  v4i gz = {0, 0, 0, 0};
#if defined(__clang_major__) && (__clang_major__ >= 23)
  v8i gz8 = {0, 0, 0, 0, 0, 0, 0, 0};
  __builtin_amdgcn_tensor_load_to_lds(g0, g1, gz, gz, gz8, 0);
#else
  __builtin_amdgcn_tensor_load_to_lds(g0, g1, gz, gz, 0);
#endif
}

// A fragment: 16x32 bf16; lanes 0-15 hold K={0..7,16..23}, lanes 16-31 K={8..15,24..31}
__device__ __forceinline__ v16bf load_a_frag(const __bf16 (*T)[40], int row0, int lane) {
  v16bf f;
  const int r  = row0 + (lane & 15);
  const int kb = (lane < 16) ? 0 : 8;
#pragma unroll
  for (int p = 0; p < 8; ++p) {
    f[p]     = T[r][kb + p];
    f[p + 8] = T[r][16 + kb + p];
  }
  return f;
}

// B fragment (W rows = output cols): lanes 0-15 hold K={0..7,16..23}, lanes 16-31 K={8..15,24..31}, N = lane&15
__device__ __forceinline__ v16bf load_b_frag(const __bf16 (*T)[40], int row0, int lane) {
  v16bf f;
  const int r  = row0 + (lane & 15);
  const int kh = (lane < 16) ? 0 : 8;
#pragma unroll
  for (int v = 0; v < 8; ++v) {
    f[v]     = T[r][kh + v];
    f[v + 8] = T[r][16 + kh + v];
  }
  return f;
}

// ----------------------------- WMMA GEMM -----------------------------------
// D[M,N] (f32) = act( A[M,K](bf16, row stride lda) * W[N,K](bf16, packed)^T + bias )
template <int ACT, bool USE_TDM>
__global__ __launch_bounds__(256) void wmma_linear_kernel(
    const __bf16* __restrict__ A, int lda,
    const __bf16* __restrict__ W,
    const float* __restrict__ bias,
    float* __restrict__ D, int ldd,
    int M, int N, int K) {
  __shared__ __bf16 As[2][128][40];
  __shared__ __bf16 Ws[2][64][40];

  const int tid  = threadIdx.x;
  const int lane = tid & 31;
  const int wid  = tid >> 5;
  const int wm   = wid >> 1;            // 4x2 wave grid; wave sub-tile 32x32
  const int wn   = wid & 1;
  const int m0   = blockIdx.y * 128;
  const int n0   = blockIdx.x * 64;
  const int ktiles = (K + 31) / 32;

  v8f acc[2][2] = {};

  if (USE_TDM && wid == 0) {  // prologue: stage K-tile 0 into buffer 0
    tdm_load_tile((unsigned)(size_t)&As[0][0][0], A + (size_t)m0 * lda,
                  M - m0, K, (long long)lda, 128u);
    tdm_load_tile((unsigned)(size_t)&Ws[0][0][0], W + (size_t)n0 * K,
                  N - n0, K, (long long)K, 64u);
  }

  for (int kt = 0; kt < ktiles; ++kt) {
    int cb;  // compute buffer
    if (USE_TDM) {
      cb = kt & 1;
      if (wid == 0) {
        if (kt + 1 < ktiles) {  // stream next tile behind the WMMAs
          const int k1 = (kt + 1) * 32;
          tdm_load_tile((unsigned)(size_t)&As[cb ^ 1][0][0],
                        A + (size_t)m0 * lda + k1, M - m0, K - k1,
                        (long long)lda, 128u);
          tdm_load_tile((unsigned)(size_t)&Ws[cb ^ 1][0][0],
                        W + (size_t)n0 * K + k1, N - n0, K - k1,
                        (long long)K, 64u);
          __builtin_amdgcn_s_wait_tensorcnt(2);  // current tile done, next in flight
        } else {
          __builtin_amdgcn_s_wait_tensorcnt(0);
        }
      }
      __syncthreads();
    } else {
      cb = 0;
      const int k0 = kt * 32;
      for (int i = tid; i < 128 * 32; i += 256) {
        int r = i >> 5, c = i & 31;
        int gm = m0 + r, gk = k0 + c;
        As[0][r][c] = (gm < M && gk < K) ? A[(size_t)gm * lda + gk] : (__bf16)0.f;
      }
      for (int i = tid; i < 64 * 32; i += 256) {
        int r = i >> 5, c = i & 31;
        int gn = n0 + r, gk = k0 + c;
        Ws[0][r][c] = (gn < N && gk < K) ? W[(size_t)gn * K + gk] : (__bf16)0.f;
      }
      __syncthreads();
    }

    const v16bf a0 = load_a_frag(As[cb], wm * 32, lane);
    const v16bf a1 = load_a_frag(As[cb], wm * 32 + 16, lane);
    const v16bf b0 = load_b_frag(Ws[cb], wn * 32, lane);
    const v16bf b1 = load_b_frag(Ws[cb], wn * 32 + 16, lane);

    acc[0][0] = __builtin_amdgcn_wmma_f32_16x16x32_bf16(false, a0, false, b0,
                                                        (short)0, acc[0][0], false, false);
    acc[0][1] = __builtin_amdgcn_wmma_f32_16x16x32_bf16(false, a0, false, b1,
                                                        (short)0, acc[0][1], false, false);
    acc[1][0] = __builtin_amdgcn_wmma_f32_16x16x32_bf16(false, a1, false, b0,
                                                        (short)0, acc[1][0], false, false);
    acc[1][1] = __builtin_amdgcn_wmma_f32_16x16x32_bf16(false, a1, false, b1,
                                                        (short)0, acc[1][1], false, false);
    __syncthreads();
  }

  // C/D layout: lane n (0-15) col = n, rows r (+8 for upper lane half)
#pragma unroll
  for (int i = 0; i < 2; ++i) {
#pragma unroll
    for (int j = 0; j < 2; ++j) {
      const int cn    = n0 + wn * 32 + j * 16 + (lane & 15);
      const int mbase = m0 + wm * 32 + i * 16 + ((lane < 16) ? 0 : 8);
      const float bv  = (bias != nullptr && cn < N) ? bias[cn] : 0.f;
#pragma unroll
      for (int r = 0; r < 8; ++r) {
        int cm = mbase + r;
        if (cm < M && cn < N) D[(size_t)cm * ldd + cn] = act_apply<ACT>(acc[i][j][r] + bv);
      }
    }
  }
}

// --------------------------- pointwise kernels ------------------------------

__global__ void cast_f32_bf16_kernel(const float* __restrict__ s, __bf16* __restrict__ d, size_t n) {
  for (size_t i = (size_t)blockIdx.x * blockDim.x + threadIdx.x; i < n;
       i += (size_t)gridDim.x * blockDim.x)
    d[i] = (__bf16)s[i];
}

__global__ void neg_exp_kernel(const float* __restrict__ s, float* __restrict__ d, size_t n) {
  for (size_t i = (size_t)blockIdx.x * blockDim.x + threadIdx.x; i < n;
       i += (size_t)gridDim.x * blockDim.x)
    d[i] = -__expf(s[i]);
}

__global__ void add_kernel(float* __restrict__ x, const float* __restrict__ h, size_t n) {
  for (size_t i = (size_t)blockIdx.x * blockDim.x + threadIdx.x; i < n;
       i += (size_t)gridDim.x * blockDim.x)
    x[i] += h[i];
}

// depthwise causal conv (K=4) + SiLU; input rows have stride ldx (xc slice of xz)
__global__ void dwconv_silu_kernel(const float* __restrict__ x, int ldx,
                                   const float* __restrict__ w, const float* __restrict__ b,
                                   float* __restrict__ out, int Bn, int L, int C) {
  size_t total = (size_t)Bn * L * C;
  for (size_t i = (size_t)blockIdx.x * blockDim.x + threadIdx.x; i < total;
       i += (size_t)gridDim.x * blockDim.x) {
    int c = (int)(i % C);
    size_t r = i / C;
    int l = (int)(r % L);
    int bb = (int)(r / L);
    float s = b[c];
#pragma unroll
    for (int k = 0; k < 4; ++k) {
      int lt = l - 3 + k;
      if (lt >= 0) s = fmaf(x[((size_t)bb * L + lt) * ldx + c], w[c * 4 + k], s);
    }
    out[i] = s / (1.f + __expf(-s));
  }
}

// y *= silu(z), z strided
__global__ void silu_mul_kernel(const float* __restrict__ y, const float* __restrict__ z, int ldz,
                                float* __restrict__ out, size_t rows, int C) {
  size_t total = rows * C;
  for (size_t i = (size_t)blockIdx.x * blockDim.x + threadIdx.x; i < total;
       i += (size_t)gridDim.x * blockDim.x) {
    size_t r = i / C;
    int c = (int)(i % C);
    float zv = z[r * ldz + c];
    out[i] = y[i] * (zv / (1.f + __expf(-zv)));
  }
}

// sequence reverse per batch (f32), supports strided in/out (for concat halves)
__global__ void flip_rows_kernel(const float* __restrict__ in, int ldin,
                                 float* __restrict__ out, int ldout,
                                 int Bn, int L, int C) {
  size_t total = (size_t)Bn * L * C;
  for (size_t i = (size_t)blockIdx.x * blockDim.x + threadIdx.x; i < total;
       i += (size_t)gridDim.x * blockDim.x) {
    int c = (int)(i % C);
    size_t r = i / C;
    int l = (int)(r % L);
    int bb = (int)(r / L);
    out[((size_t)bb * L + l) * ldout + c] = in[((size_t)bb * L + (L - 1 - l)) * ldin + c];
  }
}

// selective scan: one lane per (batch, channel); 16-state recurrence in registers
__global__ void scan_kernel(const float* __restrict__ u, const float* __restrict__ dt,
                            const float* __restrict__ Aneg, const float* __restrict__ Bc,
                            const float* __restrict__ Cc, int ldbc,
                            const float* __restrict__ Dv, float* __restrict__ y,
                            int Bn, int L, int C) {
  int i = blockIdx.x * blockDim.x + threadIdx.x;
  if (i >= Bn * C) return;
  int b = i / C, c = i % C;
  float Ar[16], h[16];
#pragma unroll
  for (int n = 0; n < 16; ++n) { Ar[n] = Aneg[(size_t)c * 16 + n]; h[n] = 0.f; }
  const float dcoef = Dv[c];
  for (int t = 0; t < L; ++t) {
    size_t uix = ((size_t)b * L + t) * C + c;
    size_t bix = ((size_t)b * L + t) * ldbc;
    float dtv = dt[uix], uv = u[uix];
    float du = dtv * uv;
    if (t + 1 < L) {
      __builtin_prefetch(&Bc[bix + ldbc], 0, 1);
      __builtin_prefetch(&Cc[bix + ldbc], 0, 1);
    }
    float acc = 0.f;
#pragma unroll
    for (int n = 0; n < 16; ++n) {
      h[n] = __expf(dtv * Ar[n]) * h[n] + du * Bc[bix + n];
      acc = fmaf(h[n], Cc[bix + n], acc);
    }
    y[uix] = acc + uv * dcoef;
  }
}

// LayerNorm, one wave32 per row (n <= 512)
__global__ __launch_bounds__(128) void ln_kernel(const float* __restrict__ x,
                                                 const float* __restrict__ g,
                                                 const float* __restrict__ b,
                                                 float* __restrict__ out, int rows, int n) {
  int wid  = (int)((blockIdx.x * blockDim.x + threadIdx.x) >> 5);
  int lane = threadIdx.x & 31;
  if (wid >= rows) return;
  const float* xr = x + (size_t)wid * n;
  float vals[16];
  int cnt = 0;
  float s = 0.f;
  for (int i = lane; i < n; i += 32) { float v = xr[i]; vals[cnt++] = v; s += v; }
  for (int m = 16; m; m >>= 1) s += __shfl_xor(s, m, 32);
  float mu = s / n;
  float v2 = 0.f;
  for (int j = 0; j < cnt; ++j) { float d = vals[j] - mu; v2 += d * d; }
  for (int m = 16; m; m >>= 1) v2 += __shfl_xor(v2, m, 32);
  float rstd = rsqrtf(v2 / n + 1e-5f);
  float* orow = out + (size_t)wid * n;
  int i = lane;
  for (int j = 0; j < cnt; ++j, i += 32) orow[i] = (vals[j] - mu) * rstd * g[i] + b[i];
}

// cluster-context: sim = l2norm(proj).l2norm(centers); out = xn + alpha * softmax(sim)@centers
__global__ __launch_bounds__(128) void cluster_ctx_kernel(
    const float* __restrict__ xn, const float* __restrict__ proj,
    const float* __restrict__ centers, const float* __restrict__ alpha,
    float* __restrict__ out, int rows, int D, int NC) {
  int wid  = (int)((blockIdx.x * blockDim.x + threadIdx.x) >> 5);
  int lane = threadIdx.x & 31;
  if (wid >= rows) return;
  const int per = D / 32;  // 8 for D=256
  const float* pr = proj + (size_t)wid * D;
  float pv[8];
  float n2 = 0.f;
  for (int j = 0; j < per; ++j) { float v = pr[lane + 32 * j]; pv[j] = v; n2 += v * v; }
  for (int m = 16; m; m >>= 1) n2 += __shfl_xor(n2, m, 32);
  float pn = fmaxf(sqrtf(n2), 1e-12f);
  float sim[8];
  float smax = -1e30f;
  for (int c = 0; c < NC; ++c) {
    const float* cr = centers + (size_t)c * D;
    float d = 0.f, cn = 0.f;
    for (int j = 0; j < per; ++j) {
      float cv = cr[lane + 32 * j];
      d += pv[j] * cv;
      cn += cv * cv;
    }
    for (int m = 16; m; m >>= 1) { d += __shfl_xor(d, m, 32); cn += __shfl_xor(cn, m, 32); }
    sim[c] = d / (pn * fmaxf(sqrtf(cn), 1e-12f));
    smax = fmaxf(smax, sim[c]);
  }
  float ssum = 0.f;
  for (int c = 0; c < NC; ++c) { sim[c] = __expf(sim[c] - smax); ssum += sim[c]; }
  float inv = 1.f / ssum;
  float a = alpha[0];
  const float* xr = xn + (size_t)wid * D;
  float* orow = out + (size_t)wid * D;
  for (int j = 0; j < per; ++j) {
    int d = lane + 32 * j;
    float ctx = 0.f;
    for (int c = 0; c < NC; ++c) ctx += sim[c] * inv * centers[(size_t)c * D + d];
    orow[d] = xr[d] + a * ctx;
  }
}

// x += softmax2(logits)[0]*mo + [1]*cc
__global__ void gate_combine_kernel(float* __restrict__ x, const float* __restrict__ logits,
                                    const float* __restrict__ mo, const float* __restrict__ cc,
                                    size_t rows, int D) {
  size_t total = rows * D;
  for (size_t i = (size_t)blockIdx.x * blockDim.x + threadIdx.x; i < total;
       i += (size_t)gridDim.x * blockDim.x) {
    size_t r = i / D;
    float l0 = logits[r * 2], l1 = logits[r * 2 + 1];
    float m = fmaxf(l0, l1);
    float e0 = __expf(l0 - m), e1 = __expf(l1 - m);
    float inv = 1.f / (e0 + e1);
    x[i] += e0 * inv * mo[i] + e1 * inv * cc[i];
  }
}

// concat([mean over L, max over L]) -> [Bn, 2D]
__global__ void pool_kernel(const float* __restrict__ f, float* __restrict__ out,
                            int Bn, int L, int D) {
  int i = blockIdx.x * blockDim.x + threadIdx.x;
  if (i >= Bn * D) return;
  int b = i / D, d = i % D;
  float s = 0.f, mx = -1e30f;
  for (int l = 0; l < L; ++l) {
    float v = f[((size_t)b * L + l) * D + d];
    s += v;
    mx = fmaxf(mx, v);
  }
  out[(size_t)b * 2 * D + d]     = s / L;
  out[(size_t)b * 2 * D + D + d] = mx;
}

// ----------------------------- CNN kernels ----------------------------------

__global__ void conv1d_kernel(const float* __restrict__ x, const float* __restrict__ w,
                              const float* __restrict__ bias, float* __restrict__ out,
                              int Bn, int Cin, int Lin, int Cout, int Kw, int stride, int pad,
                              int Lout, int co_off, int Ctot) {
  size_t total = (size_t)Bn * Cout * Lout;
  for (size_t i = (size_t)blockIdx.x * blockDim.x + threadIdx.x; i < total;
       i += (size_t)gridDim.x * blockDim.x) {
    int l = (int)(i % Lout);
    size_t r = i / Lout;
    int co = (int)(r % Cout);
    int b = (int)(r / Cout);
    float s = bias ? bias[co] : 0.f;
    int l0 = l * stride - pad;
    for (int ci = 0; ci < Cin; ++ci) {
      const float* xp = x + ((size_t)b * Cin + ci) * Lin;
      const float* wp = w + ((size_t)co * Cin + ci) * Kw;
      for (int k = 0; k < Kw; ++k) {
        int li = l0 + k;
        if (li >= 0 && li < Lin) s = fmaf(xp[li], wp[k], s);
      }
    }
    out[((size_t)b * Ctot + co_off + co) * Lout + l] = s;
  }
}

__global__ void bn_act_kernel(float* __restrict__ x, const float* __restrict__ g,
                              const float* __restrict__ b, const float* __restrict__ m,
                              const float* __restrict__ v, int C, int L, size_t total,
                              int do_silu) {
  for (size_t i = (size_t)blockIdx.x * blockDim.x + threadIdx.x; i < total;
       i += (size_t)gridDim.x * blockDim.x) {
    int c = (int)((i / L) % C);
    float scale = g[c] * rsqrtf(v[c] + 1e-5f);
    float val = (x[i] - m[c]) * scale + b[c];
    x[i] = do_silu ? val / (1.f + __expf(-val)) : val;
  }
}

__global__ void maxpool3_kernel(const float* __restrict__ x, float* __restrict__ out,
                                int Bn, int C, int L) {
  size_t total = (size_t)Bn * C * L;
  for (size_t i = (size_t)blockIdx.x * blockDim.x + threadIdx.x; i < total;
       i += (size_t)gridDim.x * blockDim.x) {
    int l = (int)(i % L);
    size_t rc = i / L;
    const float* xp = x + rc * L;
    float v = xp[l];
    if (l > 0) v = fmaxf(v, xp[l - 1]);
    if (l + 1 < L) v = fmaxf(v, xp[l + 1]);
    out[i] = v;
  }
}

__global__ void transpose_bcl_blc_kernel(const float* __restrict__ x, float* __restrict__ out,
                                         int Bn, int C, int L) {
  size_t total = (size_t)Bn * C * L;
  for (size_t i = (size_t)blockIdx.x * blockDim.x + threadIdx.x; i < total;
       i += (size_t)gridDim.x * blockDim.x) {
    int l = (int)(i % L);
    size_t r = i / L;
    int c = (int)(r % C);
    int b = (int)(r / C);
    out[((size_t)b * L + l) * C + c] = x[i];
  }
}

// ------------------------------ host side -----------------------------------

static inline dim3 gs(size_t n, int blk = 256) {
  size_t g = (n + blk - 1) / blk;
  if (g > 1048576) g = 1048576;
  if (g == 0) g = 1;
  return dim3((unsigned)g);
}

static void gemm_launch(const __bf16* A, int lda, const __bf16* W, const float* bias,
                        float* D, int ldd, int M, int N, int K, int act, hipStream_t s) {
  dim3 grid((N + 63) / 64, (M + 127) / 128), blk(256);
  const bool tdm = (K >= 32);  // K=16 (dt projection) uses the manual-copy path
#define GL(ACT, TDM) \
  wmma_linear_kernel<ACT, TDM><<<grid, blk, 0, s>>>(A, lda, W, bias, D, ldd, M, N, K)
  if (tdm) {
    switch (act) {
      case 0: GL(0, true); break;
      case 1: GL(1, true); break;
      case 2: GL(2, true); break;
      default: GL(3, true); break;
    }
  } else {
    switch (act) {
      case 0: GL(0, false); break;
      case 1: GL(1, false); break;
      case 2: GL(2, false); break;
      default: GL(3, false); break;
    }
  }
#undef GL
}

extern "C" void kernel_launch(void* const* d_in, const int* in_sizes, int n_in,
                              void* d_out, int out_size, void* d_ws, size_t ws_size,
                              hipStream_t stream) {
  (void)in_sizes; (void)n_in; (void)out_size; (void)ws_size;
  const int Bn = 32, DM = 256, DI = 512, L = 512, NC = 8;
  const int BL = Bn * L;  // 16384 tokens

  // jax pytree flatten order (sorted dict keys; lists in order):
  //   [0..215]  blocks: 6 x 36 leaves
  //     +0 alpha | +1 cc_norm.b +2 cc_norm.g | +3 cc_proj.b +4 cc_proj.w | +5 centers
  //     +6 ffn1.b +7 ffn1.w | +8 ffn2.b +9 ffn2.w | +10 ffn_ln.b +11 ffn_ln.g
  //     +12 fusion.b +13 fusion.w | +14 gate.b +15 gate.w
  //     +16..24 mb {A_log,D,W_dt,W_in,W_out,W_x,b_dt,conv_b,conv_w}
  //     +25..33 mf {same} | +34 norm1.b +35 norm1.g
  //   [216..225] cls1.b/w cls2.b/w cls3.b/w cls_ln1.b/g cls_ln2.b/g
  //   [226..265] cnn (down_bn b/g/m/v, down_w, inc1 x9, inc2 x9, proj_b, proj_w,
  //              res_bn x4, res_w, stem_bn1 x4, stem_bn2 x4, stem_w1, stem_w2)
  //   [266..267] final_norm.b/g ; [268] x
  auto F = [&](int i) { return (const float*)d_in[i]; };
  const float* x_in = F(268);

  // workspace bump allocator
  char* wsb = (char*)d_ws;
  size_t off = 0;
  auto allocf = [&](size_t n) -> float* {
    off = (off + 255) & ~(size_t)255;
    float* r = (float*)(wsb + off);
    off += n * sizeof(float);
    return r;
  };
  auto allocb = [&](size_t n) -> __bf16* {
    off = (off + 255) & ~(size_t)255;
    __bf16* r = (__bf16*)(wsb + off);
    off += n * sizeof(__bf16);
    return r;
  };

  auto cast16 = [&](const float* s, __bf16* d, size_t n) {
    cast_f32_bf16_kernel<<<gs(n), 256, 0, stream>>>(s, d, n);
  };
  auto conv = [&](const float* xp, const float* wp, const float* bp, float* op, int Cin,
                  int Lin, int Cout, int Kw, int stride, int pad, int Lout, int co_off,
                  int Ctot) {
    conv1d_kernel<<<gs((size_t)Bn * Cout * Lout), 256, 0, stream>>>(
        xp, wp, bp, op, Bn, Cin, Lin, Cout, Kw, stride, pad, Lout, co_off, Ctot);
  };
  auto bnact = [&](float* xp, int pidx, int C, int Lx, int silu) {
    bn_act_kernel<<<gs((size_t)Bn * C * Lx), 256, 0, stream>>>(
        xp, F(pidx + 1), F(pidx + 0), F(pidx + 2), F(pidx + 3), C, Lx, (size_t)Bn * C * Lx, silu);
  };
  auto lnorm = [&](const float* xp, int bidx, int gidx, float* op, int rows, int n) {
    ln_kernel<<<gs((size_t)rows * 32, 128), 128, 0, stream>>>(xp, F(gidx), F(bidx), op, rows, n);
  };

  // persistent feature buffer f [BL, 256]
  float* f = allocf((size_t)BL * DM);

  // ---------------- CNN extractor (temps freed after use) ----------------
  {
    size_t mark = off;
    float* t0   = allocf((size_t)Bn * 64 * 2048);
    float* t1   = allocf((size_t)Bn * 128 * 1024);
    float* resb = allocf((size_t)Bn * 192 * 1024);
    float* cat  = allocf((size_t)Bn * 192 * 1024);
    float* bott = allocf((size_t)Bn * 32 * 1024);
    float* mpb  = allocf((size_t)Bn * 192 * 1024);
    float* dn   = allocf((size_t)Bn * 192 * 512);
    float* cat2 = allocf((size_t)Bn * 256 * 512);
    float* pr   = allocf((size_t)Bn * 256 * 512);

    conv(x_in, F(264), nullptr, t0, 12, 4096, 64, 15, 2, 7, 2048, 0, 64);
    bnact(t0, 256, 64, 2048, 1);
    conv(t0, F(265), nullptr, t1, 64, 2048, 128, 15, 2, 7, 1024, 0, 128);
    bnact(t1, 260, 128, 1024, 1);
    conv(t1, F(255), nullptr, resb, 128, 1024, 192, 1, 1, 0, 1024, 0, 192);
    bnact(resb, 251, 192, 1024, 0);

    auto inception = [&](const float* xin, int Cin, int Lx, int nf, int pbase, float* catb) {
      conv(xin, F(pbase + 4), nullptr, bott, Cin, Lx, 32, 1, 1, 0, Lx, 0, 32);
      conv(bott, F(pbase + 5), nullptr, catb, 32, Lx, nf, 5, 1, 2, Lx, 0, 4 * nf);
      conv(bott, F(pbase + 6), nullptr, catb, 32, Lx, nf, 11, 1, 5, Lx, nf, 4 * nf);
      conv(bott, F(pbase + 7), nullptr, catb, 32, Lx, nf, 23, 1, 11, Lx, 2 * nf, 4 * nf);
      maxpool3_kernel<<<gs((size_t)Bn * Cin * Lx), 256, 0, stream>>>(xin, mpb, Bn, Cin, Lx);
      conv(mpb, F(pbase + 8), nullptr, catb, Cin, Lx, nf, 1, 1, 0, Lx, 3 * nf, 4 * nf);
      bnact(catb, pbase + 0, 4 * nf, Lx, 1);
    };
    inception(t1, 128, 1024, 48, 231, cat);
    add_kernel<<<gs((size_t)Bn * 192 * 1024), 256, 0, stream>>>(cat, resb, (size_t)Bn * 192 * 1024);
    conv(cat, F(230), nullptr, dn, 192, 1024, 192, 3, 2, 1, 512, 0, 192);
    bnact(dn, 226, 192, 512, 1);
    inception(dn, 192, 512, 64, 240, cat2);
    conv(cat2, F(250), F(249), pr, 256, 512, 256, 1, 1, 0, 512, 0, 256);
    transpose_bcl_blc_kernel<<<gs((size_t)Bn * 256 * 512), 256, 0, stream>>>(pr, f, Bn, 256, 512);
    off = mark;  // release CNN temporaries
  }

  // ---------------- Mamba block working set ----------------
  float* xn      = allocf((size_t)BL * DM);
  float* xnflip  = allocf((size_t)BL * DM);
  float* xz      = allocf((size_t)BL * 2 * DI);   // also reused for ffn1 output
  float* convout = allocf((size_t)BL * DI);
  float* xdb     = allocf((size_t)BL * 48);
  float* dtbuf   = allocf((size_t)BL * DI);
  float* Aneg    = allocf((size_t)DI * 16);
  float* ybuf    = allocf((size_t)BL * DI);
  float* sil     = allocf((size_t)BL * DI);
  float* concat  = allocf((size_t)BL * 2 * DM);
  float* mtmp    = allocf((size_t)BL * DM);
  float* mamba_o = allocf((size_t)BL * DM);
  float* projb   = allocf((size_t)BL * DM);
  float* ccx     = allocf((size_t)BL * DM);
  float* ccout   = allocf((size_t)BL * DM);
  float* gateb   = allocf((size_t)BL * 2);
  float* ffn2o   = allocf((size_t)BL * DM);

  __bf16* xn16     = allocb((size_t)BL * DM);
  __bf16* xnflip16 = allocb((size_t)BL * DM);
  __bf16* conv16   = allocb((size_t)BL * DI);
  __bf16* xdb16    = allocb((size_t)BL * 48);
  __bf16* sil16    = allocb((size_t)BL * DI);
  __bf16* concat16 = allocb((size_t)BL * 2 * DM);
  __bf16* big16    = allocb((size_t)BL * 2 * DI);
  __bf16* wbuf16   = allocb((size_t)1024 * 256);  // largest weight, reused serially

  auto run_mamba = [&](const __bf16* xin16, int mp /*leaf base*/, float* outp, int ldout) {
    // mp: +0 A_log +1 D +2 W_dt +3 W_in +4 W_out +5 W_x +6 b_dt +7 conv_b +8 conv_w
    cast16(F(mp + 3), wbuf16, (size_t)2 * DI * DM);
    gemm_launch(xin16, DM, wbuf16, nullptr, xz, 2 * DI, BL, 2 * DI, DM, 0, stream);
    dwconv_silu_kernel<<<gs((size_t)BL * DI), 256, 0, stream>>>(
        xz, 2 * DI, F(mp + 8), F(mp + 7), convout, Bn, L, DI);
    cast16(convout, conv16, (size_t)BL * DI);
    cast16(F(mp + 5), wbuf16, (size_t)48 * DI);
    gemm_launch(conv16, DI, wbuf16, nullptr, xdb, 48, BL, 48, DI, 0, stream);
    cast16(xdb, xdb16, (size_t)BL * 48);
    cast16(F(mp + 2), wbuf16, (size_t)DI * 16);
    gemm_launch(xdb16, 48, wbuf16, F(mp + 6), dtbuf, DI, BL, DI, 16, 3, stream);  // softplus
    neg_exp_kernel<<<gs((size_t)DI * 16), 256, 0, stream>>>(F(mp + 0), Aneg, (size_t)DI * 16);
    scan_kernel<<<gs((size_t)Bn * DI), 256, 0, stream>>>(convout, dtbuf, Aneg, xdb + 16,
                                                         xdb + 32, 48, F(mp + 1), ybuf, Bn, L, DI);
    silu_mul_kernel<<<gs((size_t)BL * DI), 256, 0, stream>>>(ybuf, xz + DI, 2 * DI, sil,
                                                             (size_t)BL, DI);
    cast16(sil, sil16, (size_t)BL * DI);
    cast16(F(mp + 4), wbuf16, (size_t)DM * DI);
    gemm_launch(sil16, DI, wbuf16, nullptr, outp, ldout, BL, DM, DI, 0, stream);
  };

  for (int blk = 0; blk < 6; ++blk) {
    const int pb = blk * 36;
    lnorm(f, pb + 34, pb + 35, xn, BL, DM);
    cast16(xn, xn16, (size_t)BL * DM);

    // forward mamba -> concat[:, :256]
    run_mamba(xn16, pb + 25, concat, 2 * DM);
    // backward mamba: flip input, run, flip output into concat[:, 256:]
    flip_rows_kernel<<<gs((size_t)BL * DM), 256, 0, stream>>>(xn, DM, xnflip, DM, Bn, L, DM);
    cast16(xnflip, xnflip16, (size_t)BL * DM);
    run_mamba(xnflip16, pb + 16, mtmp, DM);
    flip_rows_kernel<<<gs((size_t)BL * DM), 256, 0, stream>>>(mtmp, DM, concat + DM, 2 * DM,
                                                              Bn, L, DM);
    // fusion
    cast16(concat, concat16, (size_t)BL * 2 * DM);
    cast16(F(pb + 13), wbuf16, (size_t)DM * 2 * DM);
    gemm_launch(concat16, 2 * DM, wbuf16, F(pb + 12), mamba_o, DM, BL, DM, 2 * DM, 0, stream);

    // cluster-context path
    cast16(F(pb + 4), wbuf16, (size_t)DM * DM);
    gemm_launch(xn16, DM, wbuf16, F(pb + 3), projb, DM, BL, DM, DM, 0, stream);
    cluster_ctx_kernel<<<gs((size_t)BL * 32, 128), 128, 0, stream>>>(
        xn, projb, F(pb + 5), F(pb + 0), ccx, BL, DM, NC);
    lnorm(ccx, pb + 1, pb + 2, ccout, BL, DM);

    // gate + residual combine
    cast16(F(pb + 15), wbuf16, (size_t)2 * DM);
    gemm_launch(xn16, DM, wbuf16, F(pb + 14), gateb, 2, BL, 2, DM, 0, stream);
    gate_combine_kernel<<<gs((size_t)BL * DM), 256, 0, stream>>>(f, gateb, mamba_o, ccout,
                                                                 (size_t)BL, DM);
    // FFN
    lnorm(f, pb + 10, pb + 11, xn, BL, DM);
    cast16(xn, xn16, (size_t)BL * DM);
    cast16(F(pb + 7), wbuf16, (size_t)4 * DM * DM);
    gemm_launch(xn16, DM, wbuf16, F(pb + 6), xz, 4 * DM, BL, 4 * DM, DM, 2, stream);  // GELU
    cast16(xz, big16, (size_t)BL * 4 * DM);
    cast16(F(pb + 9), wbuf16, (size_t)DM * 4 * DM);
    gemm_launch(big16, 4 * DM, wbuf16, F(pb + 8), ffn2o, DM, BL, DM, 4 * DM, 0, stream);
    add_kernel<<<gs((size_t)BL * DM), 256, 0, stream>>>(f, ffn2o, (size_t)BL * DM);
  }

  // ---------------- head ----------------
  lnorm(f, 266, 267, xn, BL, DM);
  float* pooled = allocf((size_t)Bn * 2 * DM);
  float* c1  = allocf((size_t)Bn * DM);
  float* c1n = allocf((size_t)Bn * DM);
  float* c2  = allocf((size_t)Bn * 128);
  float* c2n = allocf((size_t)Bn * 128);
  __bf16* pooled16 = allocb((size_t)Bn * 2 * DM);
  __bf16* c1n16 = allocb((size_t)Bn * DM);
  __bf16* c2n16 = allocb((size_t)Bn * 128);

  pool_kernel<<<gs((size_t)Bn * DM), 256, 0, stream>>>(xn, pooled, Bn, L, DM);
  cast16(pooled, pooled16, (size_t)Bn * 2 * DM);
  cast16(F(217), wbuf16, (size_t)DM * 2 * DM);
  gemm_launch(pooled16, 2 * DM, wbuf16, F(216), c1, DM, Bn, DM, 2 * DM, 2, stream);
  lnorm(c1, 222, 223, c1n, Bn, DM);
  cast16(c1n, c1n16, (size_t)Bn * DM);
  cast16(F(219), wbuf16, (size_t)128 * DM);
  gemm_launch(c1n16, DM, wbuf16, F(218), c2, 128, Bn, 128, DM, 2, stream);
  lnorm(c2, 224, 225, c2n, Bn, 128);
  cast16(c2n, c2n16, (size_t)Bn * 128);
  cast16(F(221), wbuf16, (size_t)5 * 128);
  gemm_launch(c2n16, 128, wbuf16, F(220), (float*)d_out, 5, Bn, 5, 128, 0, stream);
}
